// MLSTMBlock_42039139893344
// MI455X (gfx1250) — matrix operand order repaired and checked by reading, not compiled, not run on hardware
//
#include <hip/hip_runtime.h>
#include <hip/hip_bf16.h>

typedef __attribute__((ext_vector_type(16))) _Float16 v16h;
typedef __attribute__((ext_vector_type(8)))  float    v8f;

#define B_   2
#define S_   512
#define D_   128
#define D2_  256
#define HP_  256
#define ROWS_ (B_ * S_)      // 1024
#define EPS_ 1e-6f

// ---------------------------------------------------------------------------
// WMMA fragment helpers (f32 source -> f16 fragments, 16x16x32 shape)
// A 16x32 layout: lane<16 -> M=lane, halves e: K = (e<8 ? e : e+8)
//                 lane>=16 -> M=lane-16, K = 8 + (e<8 ? e : e+8)
// B 32x16 layout: same K as A: lane<16 -> K base 0, lane>=16 -> K base 8, K += (e<8 ? e : e+8); N = lane&15
// C/D 16x16:      VGPR r -> M = r + (lane<16 ? 0 : 8); N = lane&15
// ---------------------------------------------------------------------------
__device__ inline v16h load_a_frag(const float* __restrict__ A, int lda,
                                   int row0, int k0) {
  const int lane = threadIdx.x & 31;
  const int m = lane & 15;
  const int kb = (lane < 16) ? 0 : 8;
  const float* ap = A + (size_t)(row0 + m) * lda + k0;
  v16h a;
#pragma unroll
  for (int e = 0; e < 16; ++e) {
    const int K = kb + (e < 8 ? e : e + 8);
    a[e] = (_Float16)ap[K];
  }
  return a;
}

__device__ inline v16h load_b_frag(const float* __restrict__ W, int ldw,
                                   int k0, int col0) {
  const int lane = threadIdx.x & 31;
  const int n = lane & 15;
  const int kb = (lane < 16) ? 0 : 8;
  const float* wp = W + (size_t)(k0 + kb) * ldw + col0 + n;
  v16h b;
#pragma unroll
  for (int e = 0; e < 16; ++e) b[e] = (_Float16)wp[(size_t)(e < 8 ? e : e + 8) * ldw];
  return b;
}

template <int ACT>
__device__ inline float apply_act(float x) {
  if (ACT == 1) return x / (1.f + __expf(-x));   // silu
  if (ACT == 2) return 1.f / (1.f + __expf(-x)); // sigmoid
  if (ACT == 3) return __expf(x);                // exp
  return x;
}

// ---------------------------------------------------------------------------
// Generic WMMA GEMM: one wave per 16x64 output strip (4 accumulators, A
// fragment reused across 4 B fragments -> 4 independent WMMAs per K-step).
// grid = (N/64, M/16), block = 32.  bias index = col & bias_mask (pow2).
// ---------------------------------------------------------------------------
template <int ACT, bool HASMUL, bool HASADD>
__global__ void gemm_wmma_kernel(const float* __restrict__ A, int lda,
                                 const float* __restrict__ W, int ldw,
                                 const float* __restrict__ bias, int bias_mask,
                                 float post_scale,
                                 const float* __restrict__ mul,
                                 const float* __restrict__ addp, int ldadd,
                                 float* __restrict__ out, int ldo, int K) {
  const int row0 = blockIdx.y * 16;
  const int col0 = blockIdx.x * 64;
  v8f acc[4] = {v8f{}, v8f{}, v8f{}, v8f{}};
  for (int k0 = 0; k0 < K; k0 += 32) {
    const v16h a = load_a_frag(A, lda, row0, k0);
    const v16h b0 = load_b_frag(W, ldw, k0, col0);
    const v16h b1 = load_b_frag(W, ldw, k0, col0 + 16);
    const v16h b2 = load_b_frag(W, ldw, k0, col0 + 32);
    const v16h b3 = load_b_frag(W, ldw, k0, col0 + 48);
    acc[0] = __builtin_amdgcn_wmma_f32_16x16x32_f16(false, a, false, b0,
                                                    (short)0, acc[0], false, false);
    acc[1] = __builtin_amdgcn_wmma_f32_16x16x32_f16(false, a, false, b1,
                                                    (short)0, acc[1], false, false);
    acc[2] = __builtin_amdgcn_wmma_f32_16x16x32_f16(false, a, false, b2,
                                                    (short)0, acc[2], false, false);
    acc[3] = __builtin_amdgcn_wmma_f32_16x16x32_f16(false, a, false, b3,
                                                    (short)0, acc[3], false, false);
  }
  const int lane = threadIdx.x & 31;
  const int rbase = row0 + ((lane < 16) ? 0 : 8);
#pragma unroll
  for (int tt = 0; tt < 4; ++tt) {
    const int col = col0 + tt * 16 + (lane & 15);
    const float bv = bias[col & bias_mask];
#pragma unroll
    for (int r = 0; r < 8; ++r) {
      const int row = rbase + r;
      float t = (acc[tt][r] + bv) * post_scale;
      t = apply_act<ACT>(t);
      const size_t oidx = (size_t)row * ldo + col;
      if (HASMUL) t *= mul[oidx];
      if (HASADD) t += addp[(size_t)row * ldadd + col];
      out[oidx] = t;
    }
  }
}

// ---------------------------------------------------------------------------
// Width-4 "SAME" conv as a K=1024 WMMA GEMM with a gathered A tile:
// qk[b,t,:] = silu(b_conv + sum_{w=0..3} xu[b,t-1+w,:] @ W_conv[w])
// W_conv (4,256,256) is exactly a row-major 1024x256 matrix.
// ---------------------------------------------------------------------------
__device__ inline v16h load_a_frag_conv(const float* __restrict__ xu,
                                        int row0, int k0) {
  const int lane = threadIdx.x & 31;
  const int m = lane & 15;
  const int kb = (lane < 16) ? 0 : 8;
  const int row = row0 + m;
  const int b = row >> 9;       // /S_
  const int t = row & (S_ - 1);
  v16h a;
#pragma unroll
  for (int e = 0; e < 16; ++e) {
    const int K = k0 + kb + (e < 8 ? e : e + 8);
    const int w = K >> 8;       // tap
    const int i = K & 255;      // channel
    const int st = t - 1 + w;   // SAME pad: lo=1, hi=2
    float val = 0.f;
    if (st >= 0 && st < S_)
      val = xu[(((size_t)b * S_ + st) << 8) + i];
    a[e] = (_Float16)val;
  }
  return a;
}

__global__ void conv_wmma_kernel(const float* __restrict__ xu,
                                 const float* __restrict__ Wc,
                                 const float* __restrict__ bias,
                                 float* __restrict__ out) {
  const int row0 = blockIdx.y * 16;
  const int col0 = blockIdx.x * 64;
  v8f acc[4] = {v8f{}, v8f{}, v8f{}, v8f{}};
  for (int k0 = 0; k0 < 4 * D2_; k0 += 32) {
    const v16h a = load_a_frag_conv(xu, row0, k0);
    const v16h b0 = load_b_frag(Wc, D2_, k0, col0);
    const v16h b1 = load_b_frag(Wc, D2_, k0, col0 + 16);
    const v16h b2 = load_b_frag(Wc, D2_, k0, col0 + 32);
    const v16h b3 = load_b_frag(Wc, D2_, k0, col0 + 48);
    acc[0] = __builtin_amdgcn_wmma_f32_16x16x32_f16(false, a, false, b0,
                                                    (short)0, acc[0], false, false);
    acc[1] = __builtin_amdgcn_wmma_f32_16x16x32_f16(false, a, false, b1,
                                                    (short)0, acc[1], false, false);
    acc[2] = __builtin_amdgcn_wmma_f32_16x16x32_f16(false, a, false, b2,
                                                    (short)0, acc[2], false, false);
    acc[3] = __builtin_amdgcn_wmma_f32_16x16x32_f16(false, a, false, b3,
                                                    (short)0, acc[3], false, false);
  }
  const int lane = threadIdx.x & 31;
  const int rbase = row0 + ((lane < 16) ? 0 : 8);
#pragma unroll
  for (int tt = 0; tt < 4; ++tt) {
    const int col = col0 + tt * 16 + (lane & 15);
    const float bv = bias[col];
#pragma unroll
    for (int r = 0; r < 8; ++r) {
      float t = acc[tt][r] + bv;
      t = t / (1.f + __expf(-t)); // silu
      out[(size_t)(rbase + r) * D2_ + col] = t;
    }
  }
}

// ---------------------------------------------------------------------------
// LayerNorm over D=128: one wave per row.
// ---------------------------------------------------------------------------
__global__ void ln_kernel(const float* __restrict__ x,
                          const float* __restrict__ g,
                          const float* __restrict__ bb,
                          float* __restrict__ xn) {
  const int row = blockIdx.x;
  const int lane = threadIdx.x;
  const float* xr = x + (size_t)row * D_;
  float v[4], s = 0.f, sq = 0.f;
#pragma unroll
  for (int j = 0; j < 4; ++j) {
    v[j] = xr[lane + j * 32];
    s += v[j];
    sq += v[j] * v[j];
  }
#pragma unroll
  for (int off = 16; off > 0; off >>= 1) {
    s  += __shfl_xor(s,  off, 32);
    sq += __shfl_xor(sq, off, 32);
  }
  const float mu = s * (1.f / D_);
  const float var = sq * (1.f / D_) - mu * mu;
  const float rstd = rsqrtf(var + EPS_);
#pragma unroll
  for (int j = 0; j < 4; ++j) {
    const int c = lane + j * 32;
    xn[(size_t)row * D_ + c] = (v[j] - mu) * rstd * g[c] + bb[c];
  }
}

// ---------------------------------------------------------------------------
// Normalizer scan: n_t = f_t*n_{t-1} + i_t*k_t (n_{-1}=1), per channel;
// rdenom[b,t] = 1 / max(|sum_j n_t[j]*q_t[j]|, 1).  grid=B, block=HP.
// ---------------------------------------------------------------------------
__global__ void nscan_kernel(const float* __restrict__ it,
                             const float* __restrict__ ft,
                             const float* __restrict__ kb,
                             const float* __restrict__ qb,
                             float* __restrict__ rdenom) {
  const int b = blockIdx.x;
  const int j = threadIdx.x; // 0..255
  __shared__ float red[8];
  float n = 1.f;
  for (int t = 0; t < S_; ++t) {
    const size_t idx = ((size_t)(b * S_ + t) << 8) + j;
    n = ft[idx] * n + it[idx] * kb[idx];
    float val = n * qb[idx];
#pragma unroll
    for (int off = 16; off > 0; off >>= 1) val += __shfl_xor(val, off, 32);
    if ((j & 31) == 0) red[j >> 5] = val;
    __syncthreads();
    if (j == 0) {
      float sum = 0.f;
#pragma unroll
      for (int w = 0; w < 8; ++w) sum += red[w];
      rdenom[b * S_ + t] = 1.f / fmaxf(fabsf(sum), 1.f);
    }
    __syncthreads();
  }
}

// ---------------------------------------------------------------------------
// State scan: rows of C are independent. 16 WGs = 2 batches x 8 row-blocks.
// Each WG: 32 rows x 256 cols of C in registers (256 thr, 32 f32/thr).
// ht[b,t,i] = (sum_j C_t[i,j] q_t[j]) * rdenom[b,t] * ot[b,t,i]
// ---------------------------------------------------------------------------
__global__ void cscan_kernel(const float* __restrict__ ft,
                             const float* __restrict__ vb,
                             const float* __restrict__ kb,
                             const float* __restrict__ qb,
                             const float* __restrict__ ot,
                             const float* __restrict__ rdenom,
                             float* __restrict__ ht) {
  const int blk = blockIdx.x;           // 0..15
  const int b = blk >> 3;
  const int row0 = (blk & 7) * 32;
  const int tid = threadIdx.x;          // 0..255
  const int r = tid >> 3;               // row within block (0..31)
  const int cs = (tid & 7) * 32;        // column segment start
  const int row = row0 + r;
  __shared__ float kq[2 * HP_];
  float C[32];
#pragma unroll
  for (int j = 0; j < 32; ++j) C[j] = 0.f;

  for (int t = 0; t < S_; ++t) {
    const size_t base = (size_t)(b * S_ + t) << 8;
    kq[tid]        = kb[base + tid];
    kq[HP_ + tid]  = qb[base + tid];
    __syncthreads();
    const float f = ft[base + row];
    const float v = vb[base + row];
    float h = 0.f;
#pragma unroll
    for (int j = 0; j < 32; ++j) {
      const float c = f * C[j] + v * kq[cs + j];
      C[j] = c;
      h += c * kq[HP_ + cs + j];
    }
#pragma unroll
    for (int off = 4; off > 0; off >>= 1) h += __shfl_xor(h, off, 8);
    if ((tid & 7) == 0)
      ht[base + row] = h * rdenom[b * S_ + t] * ot[base + row];
    __syncthreads();
  }
}

// ---------------------------------------------------------------------------
// GroupNorm stats: per (b,h) group over (S,P) = 32768 elements. grid=8.
// ---------------------------------------------------------------------------
__global__ void gnstat_kernel(const float* __restrict__ ht,
                              float* __restrict__ stats) {
  const int g = blockIdx.x;
  const int b = g >> 2, h = g & 3;
  const int tid = threadIdx.x; // 256
  float s = 0.f, sq = 0.f;
  for (int e = tid; e < S_ * 64; e += 256) {
    const int si = e >> 6, p = e & 63;
    const float val = ht[(((size_t)(b * S_ + si)) << 8) + (h << 6) + p];
    s += val;
    sq += val * val;
  }
  __shared__ float rs[8], rq[8];
#pragma unroll
  for (int off = 16; off > 0; off >>= 1) {
    s  += __shfl_xor(s,  off, 32);
    sq += __shfl_xor(sq, off, 32);
  }
  if ((tid & 31) == 0) { rs[tid >> 5] = s; rq[tid >> 5] = sq; }
  __syncthreads();
  if (tid == 0) {
    float S = 0.f, Q = 0.f;
#pragma unroll
    for (int w = 0; w < 8; ++w) { S += rs[w]; Q += rq[w]; }
    const float mean = S * (1.f / (S_ * 64));
    const float var = Q * (1.f / (S_ * 64)) - mean * mean;
    stats[2 * g] = mean;
    stats[2 * g + 1] = rsqrtf(var + EPS_);
  }
}

// ---------------------------------------------------------------------------
// Fuse: ypre = (groupnorm(ht)*gn_scale + gn_bias + skip3) * skip2
// ---------------------------------------------------------------------------
__global__ void ypre_kernel(const float* __restrict__ ht,
                            const float* __restrict__ stats,
                            const float* __restrict__ gsc,
                            const float* __restrict__ gbi,
                            const float* __restrict__ qk,
                            const float* __restrict__ skip2,
                            float* __restrict__ ypre) {
  const size_t idx = (size_t)blockIdx.x * 256 + threadIdx.x;
  const int col = (int)(idx & 255);
  const int row = (int)(idx >> 8);
  const int b = row >> 9;
  const int h = col >> 6;
  const int g = (b << 2) | h;
  const float m = stats[2 * g], rstd = stats[2 * g + 1];
  const float hn = (ht[idx] - m) * rstd * gsc[col] + gbi[col];
  ypre[idx] = (hn + qk[idx]) * skip2[idx];
}

// ---------------------------------------------------------------------------
extern "C" void kernel_launch(void* const* d_in, const int* in_sizes, int n_in,
                              void* d_out, int out_size, void* d_ws, size_t ws_size,
                              hipStream_t stream) {
  const float* x        = (const float*)d_in[0];
  const float* ln_scale = (const float*)d_in[1];
  const float* ln_bias  = (const float*)d_in[2];
  const float* W_skip2  = (const float*)d_in[3];
  const float* b_skip2  = (const float*)d_in[4];
  const float* W_up     = (const float*)d_in[5];
  const float* b_up     = (const float*)d_in[6];
  const float* W_conv   = (const float*)d_in[7];
  const float* b_conv   = (const float*)d_in[8];
  const float* wi       = (const float*)d_in[9];
  const float* wf       = (const float*)d_in[10];
  const float* bi       = (const float*)d_in[11];
  const float* bf       = (const float*)d_in[12];
  const float* W_o      = (const float*)d_in[13];
  const float* b_o      = (const float*)d_in[14];
  const float* W_q      = (const float*)d_in[15];
  const float* b_q      = (const float*)d_in[16];
  const float* W_k      = (const float*)d_in[17];
  const float* b_k      = (const float*)d_in[18];
  const float* W_v      = (const float*)d_in[19];
  const float* b_v      = (const float*)d_in[20];
  const float* gn_scale = (const float*)d_in[21];
  const float* gn_bias  = (const float*)d_in[22];
  const float* W_out    = (const float*)d_in[23];
  const float* b_out    = (const float*)d_in[24];
  float* out = (float*)d_out;

  float* ws    = (float*)d_ws;
  float* xn    = ws;                       // 1024*128
  float* xu    = xn    + ROWS_ * D_;       // 1024*256 each below
  float* skip2 = xu    + ROWS_ * D2_;
  float* qk    = skip2 + ROWS_ * D2_;
  float* itb   = qk    + ROWS_ * D2_;
  float* ftb   = itb   + ROWS_ * D2_;
  float* otb   = ftb   + ROWS_ * D2_;
  float* qbuf  = otb   + ROWS_ * D2_;
  float* kbuf  = qbuf  + ROWS_ * D2_;
  float* vbuf  = kbuf  + ROWS_ * D2_;
  float* htb   = vbuf  + ROWS_ * D2_;
  float* ypre  = htb   + ROWS_ * D2_;
  float* rden  = ypre  + ROWS_ * D2_;      // 1024
  float* stats = rden  + ROWS_;            // 16

  const dim3 blk32(32);
  const dim3 g256(D2_ / 64, ROWS_ / 16);   // (4, 64) for N=256 outputs
  const dim3 g64h(1, ROWS_ / 16);          // (1, 64) per-head N=64
  const dim3 g128(D_ / 64, ROWS_ / 16);    // (2, 64) final N=128

  // 1) LayerNorm
  ln_kernel<<<ROWS_, 32, 0, stream>>>(x, ln_scale, ln_bias, xn);

  // 2) skip2 = silu(xn @ W_skip2 + b);  xu = xn @ W_up + b
  gemm_wmma_kernel<1, false, false><<<g256, blk32, 0, stream>>>(
      xn, D_, W_skip2, D2_, b_skip2, 255, 1.f, nullptr, nullptr, 0,
      skip2, D2_, D_);
  gemm_wmma_kernel<0, false, false><<<g256, blk32, 0, stream>>>(
      xn, D_, W_up, D2_, b_up, 255, 1.f, nullptr, nullptr, 0,
      xu, D2_, D_);

  // 3) qk = silu(conv1d_w4(xu))  (== skip3)
  conv_wmma_kernel<<<g256, blk32, 0, stream>>>(xu, W_conv, b_conv, qk);

  // 4) gates / projections
  gemm_wmma_kernel<2, false, false><<<g256, blk32, 0, stream>>>(
      xu, D2_, W_o, D2_, b_o, 255, 1.f, nullptr, nullptr, 0,
      otb, D2_, D2_);
  gemm_wmma_kernel<0, false, false><<<g256, blk32, 0, stream>>>(
      qk, D2_, W_q, D2_, b_q, 255, 1.f, nullptr, nullptr, 0,
      qbuf, D2_, D2_);
  gemm_wmma_kernel<0, false, false><<<g256, blk32, 0, stream>>>(
      qk, D2_, W_k, D2_, b_k, 255, 1.f / HP_, nullptr, nullptr, 0,
      kbuf, D2_, D2_);
  for (int h = 0; h < 4; ++h) { // it/ft: per-head 64x64 GEMM, exp epilogue
    gemm_wmma_kernel<3, false, false><<<g64h, blk32, 0, stream>>>(
        xu + h * 64, D2_, wi + h * 64, D2_, bi, 63, 1.f, nullptr, nullptr, 0,
        itb + h * 64, D2_, 64);
    gemm_wmma_kernel<3, false, false><<<g64h, blk32, 0, stream>>>(
        xu + h * 64, D2_, wf + h * 64, D2_, bf, 63, 1.f, nullptr, nullptr, 0,
        ftb + h * 64, D2_, 64);
  }
  // v = (xu @ W_v + b_v) * it
  gemm_wmma_kernel<0, true, false><<<g256, blk32, 0, stream>>>(
      xu, D2_, W_v, D2_, b_v, 255, 1.f, itb, nullptr, 0,
      vbuf, D2_, D2_);

  // 5) recurrence
  nscan_kernel<<<B_, HP_, 0, stream>>>(itb, ftb, kbuf, qbuf, rden);
  cscan_kernel<<<B_ * 8, 256, 0, stream>>>(ftb, vbuf, kbuf, qbuf, otb, rden, htb);

  // 6) group norm + fuse
  gnstat_kernel<<<8, 256, 0, stream>>>(htb, stats);
  ypre_kernel<<<ROWS_, 256, 0, stream>>>(htb, stats, gn_scale, gn_bias,
                                         qk, skip2, ypre);

  // 7) out = ypre @ W_out + b_out + x
  gemm_wmma_kernel<0, false, true><<<g128, blk32, 0, stream>>>(
      ypre, D2_, W_out, D_, b_out, 127, 1.f, nullptr, x, D_,
      out, D_, D2_);
}
